// _EmbeddingBank_51934744543352
// MI455X (gfx1250) — matrix operand order repaired and checked by reading, not compiled, not run on hardware
//
#include <hip/hip_runtime.h>
#include <hip/hip_bf16.h>

typedef __attribute__((ext_vector_type(16))) __bf16 v16bf;
typedef __attribute__((ext_vector_type(8)))  __bf16 v8bf;
typedef __attribute__((ext_vector_type(4)))  __bf16 v4bf;
typedef __attribute__((ext_vector_type(8)))  float  v8f;
typedef __attribute__((ext_vector_type(4)))  float  v4f;
typedef __attribute__((ext_vector_type(4)))  unsigned int v4u;
typedef __attribute__((ext_vector_type(8)))  int    v8i;
typedef __attribute__((ext_vector_type(4)))  int    v4i;

#define B_   1024
#define IN_  512
#define C_   128
#define HID_ 128
#define EMB_ 32
#define BM   128   // batch-row tile per block
#define KT   32    // K tile (matches wmma 16x16x32 K depth)
#define NKS  (IN_ / KT)

#if __has_include(<hip/amd_detail/amd_gfx1250_TDM.h>)
#define TDM_6ARG 1
#endif

// CDNA5 async global->LDS copy (ASYNCcnt-tracked).
__device__ __forceinline__ void async_copy_b128(unsigned lds_byte_off, const float* gaddr) {
    asm volatile("global_load_async_to_lds_b128 %0, %1, off"
                 :: "v"(lds_byte_off), "v"(gaddr)
                 : "memory");
}
__device__ __forceinline__ void wait_async_all() {
    asm volatile("s_wait_asynccnt 0x0" ::: "memory");
}

// Tensor Data Mover: one-shot 2D tile DMA global->LDS (TENSORcnt-tracked).
// Copies a [rows x cols] fp32 tile with row stride `stride` elements.
__device__ __forceinline__ void tdm_load_tile_f32(unsigned lds_byte_off, const float* gaddr,
                                                  int cols, int rows, int stride) {
    const unsigned long long ga = (unsigned long long)(size_t)gaddr;
    v4u g0;
    g0[0] = 1u;                                            // count=1 (valid user D#)
    g0[1] = lds_byte_off;                                  // lds_addr
    g0[2] = (unsigned)ga;                                  // global_addr[31:0]
    g0[3] = (unsigned)((ga >> 32) & 0x01FFFFFFu) | (2u << 30); // addr[56:32] | type=2
    v8i g1;
    g1[0] = (2 << 16);                                     // data_size = 4B
    g1[1] = (int)((IN_ & 0xFFFF) << 16);                   // tensor_dim0[15:0]
    g1[2] = (int)((IN_ >> 16) | ((B_ & 0xFFFF) << 16));    // dim0[31:16] | dim1[15:0]
    g1[3] = (int)((B_ >> 16) | (cols << 16));              // dim1[31:16] | tile_dim0
    g1[4] = rows;                                          // tile_dim1 | tile_dim2=0
    g1[5] = stride;                                        // tensor_dim0_stride[31:0]
    g1[6] = 0;                                             // stride0 hi | stride1 lo
    g1[7] = 0;                                             // stride1 hi
    v4i z4 = {0, 0, 0, 0};                                 // 2D: groups 2/3 unused
#ifdef TDM_6ARG
    v8i z8 = {0, 0, 0, 0, 0, 0, 0, 0};
    __builtin_amdgcn_tensor_load_to_lds(g0, g1, z4, z4, z8, 0);
#else
    __builtin_amdgcn_tensor_load_to_lds(g0, g1, z4, z4, 0);
#endif
}

// Fused 2-layer bank MLP:  out[b,e,c] = relu(x @ W1_c + b1_c) @ W2_c + b2_c
// One block = (class c, 128-row batch tile, bank). 256 threads = 8 waves.
__global__ __launch_bounds__(256)
void _EmbeddingBank_fused_wmma(
    const float* __restrict__ x,
    const float* __restrict__ posW1, const float* __restrict__ posB1,
    const float* __restrict__ posW2, const float* __restrict__ posB2,
    const float* __restrict__ negW1, const float* __restrict__ negB1,
    const float* __restrict__ negW2, const float* __restrict__ negB2,
    float* __restrict__ out)
{
    const int c    = blockIdx.x;   // class 0..127
    const int mt   = blockIdx.y;   // batch tile 0..7
    const int bank = blockIdx.z;   // 0 = pos, 1 = neg

    const float* W1 = bank ? negW1 : posW1;
    const float* b1 = bank ? negB1 : posB1;
    const float* W2 = bank ? negW2 : posW2;
    const float* b2 = bank ? negB2 : posB2;
    float* obase = out + (size_t)bank * ((size_t)B_ * EMB_ * C_);

    // LDS: sX 32KB(fp32,dbl, TDM dest) + sB 16KB + sH 32KB + sW2 8KB + biases
    __shared__ float  sX [2][BM * KT];    // x tile, fp32 row-major (TDM'd)
    __shared__ __bf16 sB [2][HID_ * KT];  // W1 tile, bf16 transposed [n][k]
    __shared__ __bf16 sH [BM * HID_];     // layer-1 activations bf16 [m][k]
    __shared__ __bf16 sW2[EMB_ * HID_];   // W2 bf16 transposed [e][k]
    __shared__ float  sB1[HID_];          // layer-1 bias (async-DMA'd)
    __shared__ float  sB2[EMB_];          // layer-2 bias (async-DMA'd)

    const int tid   = threadIdx.x;
    const int wave  = tid >> 5;
    const int lane  = tid & 31;
    const int lhalf = lane >> 4;   // 0 / 1
    const int l15   = lane & 15;

    const float* xrow = x  + (size_t)mt * BM * IN_;
    const float* W1c  = W1 + (size_t)c  * IN_ * HID_;

    // --- staging helpers ---
    auto issue_x_tdm = [&](int buf, int k0) {
        tdm_load_tile_f32((unsigned)(size_t)&sX[buf][0], xrow + k0, KT, BM, IN_);
    };
    // W1 tile: each thread owns 4 consecutive k rows x 4 n columns -> packed b64 LDS stores
    auto load_B_regs = [&](v4f* breg, int k0) {
        const int n4 = tid & 31;       // 32 x float4 across the 128-wide n dim
        const int kq = tid >> 5;       // 8 k-quads cover k0..k0+31
        #pragma unroll
        for (int j = 0; j < 4; ++j)
            breg[j] = *(const v4f*)(W1c + (size_t)(k0 + kq * 4 + j) * HID_ + n4 * 4);
    };
    auto store_B_lds = [&](int buf, const v4f* breg) {
        const int n4 = tid & 31;
        const int kq = tid >> 5;
        #pragma unroll
        for (int nn = 0; nn < 4; ++nn) {
            v4bf pk;
            pk[0] = (__bf16)breg[0][nn];
            pk[1] = (__bf16)breg[1][nn];
            pk[2] = (__bf16)breg[2][nn];
            pk[3] = (__bf16)breg[3][nn];
            *(v4bf*)(&sB[buf][(n4 * 4 + nn) * KT + kq * 4]) = pk;
        }
    };

    v8f acc[8] = {};   // wave's 16 rows x 128 cols of f32 accumulators

    // ---------------- prologue: stage K-tile 0 ----------------
    if (wave == 0) {
        issue_x_tdm(0, 0);                                   // TDM x tile 0
        async_copy_b128((unsigned)(size_t)&sB1[lane * 4],    // biases -> LDS (async)
                        b1 + c * HID_ + lane * 4);
        if (lane < 8)
            async_copy_b128((unsigned)(size_t)&sB2[lane * 4],
                            b2 + c * EMB_ + lane * 4);
    }
    {
        v4f breg[4];
        load_B_regs(breg, 0);
        store_B_lds(0, breg);
    }
    if (wave == 0) {
        __builtin_amdgcn_s_wait_tensorcnt(0);
        wait_async_all();
    }
    __syncthreads();

    // ---------------- Layer 1: H = relu(x @ W1_c + b1_c) ----------------
    for (int ks = 0; ks < NKS; ++ks) {
        const int  cur  = ks & 1;
        const bool more = (ks + 1) < NKS;
        v4f breg[4];
        if (more) {
            const int k0n = (ks + 1) * KT;
            if (wave == 0)
                issue_x_tdm(cur ^ 1, k0n);    // TDM next x tile
            load_B_regs(breg, k0n);           // global loads for next W1 tile
            if (ks + 2 < NKS)                 // prefetch W1 tile k+2 into cache
                __builtin_prefetch(W1c + (size_t)(ks + 2) * KT * HID_ + tid * 16, 0, 0);
        }

        // A fragment: rows wave*16..+15; lanes 0-15 hold
        // K = 0..15, lanes 16-31 hold K = 16..31 (matches B).
        const int am = wave * 16 + l15;
        const int kb = lhalf * 16;
        const float* xs = &sX[cur][am * KT];
        v4f a0 = *(const v4f*)(xs + kb);
        v4f a1 = *(const v4f*)(xs + kb + 4);
        v4f a2 = *(const v4f*)(xs + kb + 8);
        v4f a3 = *(const v4f*)(xs + kb + 12);
        v16bf afrag;
        #pragma unroll
        for (int i = 0; i < 4; ++i) {
            afrag[i]      = (__bf16)a0[i];
            afrag[i + 4]  = (__bf16)a1[i];
            afrag[i + 8]  = (__bf16)a2[i];
            afrag[i + 12] = (__bf16)a3[i];
        }

        #pragma unroll
        for (int t = 0; t < 8; ++t) {
            // B fragment: column = l15 (+tile), lanes 0-15 K=0..15, lanes 16-31 K=16..31
            const int bn = t * 16 + l15;
            v16bf bfrag = *(const v16bf*)(&sB[cur][bn * KT + lhalf * 16]);
            acc[t] = __builtin_amdgcn_wmma_f32_16x16x32_bf16(
                false, afrag, false, bfrag, (short)0, acc[t], false, false);
        }

        if (more) {
            store_B_lds(cur ^ 1, breg);       // convert + packed-b64 store next tile
            if (wave == 0)
                __builtin_amdgcn_s_wait_tensorcnt(0);  // next x tile fully in LDS
        }
        __syncthreads();
    }

    // bias + relu -> sH (bf16, row-major). C/D layout: N=l15, M=r+8*lhalf.
    #pragma unroll
    for (int t = 0; t < 8; ++t) {
        const int n = t * 16 + l15;
        const float bias = sB1[n];
        #pragma unroll
        for (int r = 0; r < 8; ++r) {
            const int m = wave * 16 + r + lhalf * 8;
            float v = acc[t][r] + bias;
            v = v > 0.f ? v : 0.f;
            sH[m * HID_ + n] = (__bf16)v;
        }
    }

    // stage W2: W2_c[k(128)][e(32)] fp32 -> bf16 transposed [e][k], packed b64 stores
    {
        const float* W2c = W2 + (size_t)c * HID_ * EMB_;
        const int e4 = tid & 7;    // 8 x float4 across the 32-wide e dim
        const int kq = tid >> 3;   // 32 k-quads cover k = 0..127
        v4f r[4];
        #pragma unroll
        for (int j = 0; j < 4; ++j)
            r[j] = *(const v4f*)(W2c + (size_t)(kq * 4 + j) * EMB_ + e4 * 4);
        #pragma unroll
        for (int nn = 0; nn < 4; ++nn) {
            v4bf pk;
            pk[0] = (__bf16)r[0][nn];
            pk[1] = (__bf16)r[1][nn];
            pk[2] = (__bf16)r[2][nn];
            pk[3] = (__bf16)r[3][nn];
            *(v4bf*)(&sW2[(e4 * 4 + nn) * HID_ + kq * 4]) = pk;
        }
    }
    __syncthreads();

    // ---------------- Layer 2: O = H @ W2_c + b2_c ----------------
    v8f acc2[2] = {};
    #pragma unroll
    for (int kk = 0; kk < 4; ++kk) {
        const int am = wave * 16 + l15;
        const int kb = kk * 32 + lhalf * 16;
        const __bf16* hs = &sH[am * HID_];
        v8bf alo = *(const v8bf*)(hs + kb);
        v8bf ahi = *(const v8bf*)(hs + kb + 8);
        v16bf afrag;
        #pragma unroll
        for (int i = 0; i < 8; ++i) { afrag[i] = alo[i]; afrag[i + 8] = ahi[i]; }

        #pragma unroll
        for (int t = 0; t < 2; ++t) {
            const int e = t * 16 + l15;
            v16bf bfrag = *(const v16bf*)(&sW2[e * HID_ + kk * 32 + lhalf * 16]);
            acc2[t] = __builtin_amdgcn_wmma_f32_16x16x32_bf16(
                false, afrag, false, bfrag, (short)0, acc2[t], false, false);
        }
    }

    // write out[b][e][c] fp32
    #pragma unroll
    for (int t = 0; t < 2; ++t) {
        const int e = t * 16 + l15;
        const float bias = sB2[e];
        #pragma unroll
        for (int r = 0; r < 8; ++r) {
            const int b = mt * BM + wave * 16 + r + lhalf * 8;
            obase[(size_t)b * EMB_ * C_ + (size_t)e * C_ + c] = acc2[t][r] + bias;
        }
    }
}

extern "C" void kernel_launch(void* const* d_in, const int* in_sizes, int n_in,
                              void* d_out, int out_size, void* d_ws, size_t ws_size,
                              hipStream_t stream) {
    const float* x     = (const float*)d_in[0];
    const float* posW1 = (const float*)d_in[1];
    const float* posB1 = (const float*)d_in[2];
    const float* posW2 = (const float*)d_in[3];
    const float* posB2 = (const float*)d_in[4];
    const float* negW1 = (const float*)d_in[5];
    const float* negB1 = (const float*)d_in[6];
    const float* negW2 = (const float*)d_in[7];
    const float* negB2 = (const float*)d_in[8];
    float* out = (float*)d_out;

    dim3 grid(C_, B_ / BM, 2);
    _EmbeddingBank_fused_wmma<<<grid, 256, 0, stream>>>(
        x, posW1, posB1, posW2, posB2, negW1, negB1, negW2, negB2, out);
}
